// TopPSearchDecoder_42623255446274
// MI455X (gfx1250) — compile-verified, matrix operand order checked by reading, not run on hardware
//
#include <hip/hip_runtime.h>
#include <hip/hip_bf16.h>

typedef __attribute__((ext_vector_type(16))) __bf16 v16bf;
typedef __attribute__((ext_vector_type(8)))  float  v8f;

#define HDIM 1024
#define SLEN 64
#define VOCAB 50257
#define NT_WO 3142           /* ceil(50257/16) */
#define VPAD  (NT_WO * 16)   /* 50272 */
#define KC_H  (HDIM / 32)    /* 32 k-chunks for K=1024 */
#define KC_2H (2 * HDIM / 32)/* 64 k-chunks for K=2048 */
#define NT_3H (3 * HDIM / 16)/* 192 n-tiles for N=3072 */
#define NT_H  (HDIM / 16)    /* 64 n-tiles for N=1024 */
#define TOPP  0.9f
#define SOS_TOKEN 1

// ---------------------------------------------------------------------------
// Pack an f32 row-major (Ktot x Ntot) matrix into bf16 WMMA B-fragment tiles.
// Tile layout: [ntile][kc][lane][j] with each lane's 16 bf16 contiguous, so a
// lane fetches its whole fragment with two global_load_b128.
// K-mapping per ISA 16-bit fragment layout:
//   lanes 0-15 : j<8 -> K=j,    j>=8 -> K=j+8   (K = 0..7, 16..23)
//   lanes16-31 : j<8 -> K=j+8,  j>=8 -> K=j+16  (K = 8..15, 24..31)
// ---------------------------------------------------------------------------
__global__ void pack_bf16_kernel(const float* __restrict__ W, __bf16* __restrict__ P,
                                 int Ktot, int Ntot, int ntiles) {
    int Kc = (Ktot + 31) >> 5;
    long long total = (long long)ntiles * Kc * 512;
    long long idx = (long long)blockIdx.x * blockDim.x + threadIdx.x;
    if (idx >= total) return;
    int r = (int)(idx & 511);
    long long t = idx >> 9;
    int kc = (int)(t % Kc);
    int ntile = (int)(t / Kc);
    int lane = r >> 4, j = r & 15;
    int col = ntile * 16 + (lane & 15);
    int kl = (lane < 16) ? ((j < 8) ? j : j + 8) : ((j < 8) ? j + 8 : j + 16);
    int k = kc * 32 + kl;
    float v = (k < Ktot && col < Ntot) ? W[(long long)k * Ntot + col] : 0.0f;
    P[idx] = (__bf16)v;
}

// Build the broadcast-A fragment staging area in LDS: for each k-chunk, 32 bf16
// (16 for each lane-half), matching the same K mapping as the packed B tiles.
__device__ inline void build_xpack(const float* x, __bf16* xp, int Kc, int tid, int nt) {
    for (int idx = tid; idx < Kc * 32; idx += nt) {
        int kc = idx >> 5, r = idx & 31;
        int half = r >> 4, j = r & 15;
        int kl = (half == 0) ? ((j < 8) ? j : j + 8) : ((j < 8) ? j + 8 : j + 16);
        xp[idx] = (__bf16)x[kc * 32 + kl];
    }
}

// One wave computes a block of TB adjacent 16-wide output tiles of
// out = x @ W (+bias). The A fragment (vector broadcast into all 16 rows) is
// loaded from LDS once per k-chunk (kc is the OUTER loop inside the tile
// block, so LICM cannot hoist all Kc fragments and spill them to scratch) and
// reused across TB independent accumulator chains.
template <int TB>
__device__ inline void wave_gemv(const __bf16* __restrict__ Wp, const __bf16* xp,
                                 int Kc, int ntiles, int Ntot,
                                 const float* __restrict__ bias, float* out,
                                 int wave, int nwaves, int lane) {
    const v16bf* ap = (const v16bf*)(xp + ((lane >> 4) << 4)); // half-select
    for (int t0 = wave * TB; t0 < ntiles; t0 += nwaves * TB) {
        v8f acc[TB];
#pragma unroll
        for (int tb = 0; tb < TB; ++tb) acc[tb] = (v8f){};
        const __bf16* bbase = Wp + (((long long)t0 * Kc) << 9) + (lane << 4);
        for (int kc = 0; kc < Kc; ++kc) {
            v16bf a = ap[kc * 2];   // one live A fragment, from LDS
#pragma unroll
            for (int tb = 0; tb < TB; ++tb) {
                if (t0 + tb < ntiles) {   // wave-uniform guard: EXEC stays full
                    v16bf b = *(const v16bf*)(bbase + (((long long)tb * Kc + kc) << 9));
                    acc[tb] = __builtin_amdgcn_wmma_f32_16x16x32_bf16(
                        false, a, false, b, (short)0, acc[tb], false, false);
                }
            }
        }
#pragma unroll
        for (int tb = 0; tb < TB; ++tb) {
            if (t0 + tb < ntiles && lane < 16) {
                int n = (t0 + tb) * 16 + lane;
                if (n < Ntot) out[n] = acc[tb][0] + (bias ? bias[n] : 0.0f);
            }
        }
    }
}

__device__ inline float sigmoidf_(float v) { return 1.0f / (1.0f + __expf(-v)); }

__device__ inline float block_sum(float v, float* red, int tid, int nt) {
    red[tid] = v; __syncthreads();
    for (int off = nt >> 1; off > 0; off >>= 1) {
        if (tid < off) red[tid] += red[tid + off];
        __syncthreads();
    }
    float r = red[0]; __syncthreads();
    return r;
}

__device__ inline float block_max(float v, float* red, int tid, int nt) {
    red[tid] = v; __syncthreads();
    for (int off = nt >> 1; off > 0; off >>= 1) {
        if (tid < off) red[tid] = fmaxf(red[tid], red[tid + off]);
        __syncthreads();
    }
    float r = red[0]; __syncthreads();
    return r;
}

__device__ inline float rand01(unsigned long long s) {
    unsigned long long z = s + 0x9E3779B97F4A7C15ull;
    z = (z ^ (z >> 30)) * 0xBF58476D1CE4E5B9ull;
    z = (z ^ (z >> 27)) * 0x94D049BB133111EBull;
    z = z ^ (z >> 31);
    return (float)(z >> 40) * (1.0f / 16777216.0f);
}

// ---------------------------------------------------------------------------
// Encoder: one persistent workgroup, 64 sequential GRU steps.
// ---------------------------------------------------------------------------
__global__ void __launch_bounds__(1024)
encoder_kernel(const int* __restrict__ tokens, const float* __restrict__ embed,
               const __bf16* __restrict__ Wxp, const __bf16* __restrict__ Whp,
               const float* __restrict__ b3, float* __restrict__ encOut) {
    __shared__ float x[HDIM], h[HDIM], gx[3 * HDIM], gh[3 * HDIM];
    __shared__ __align__(32) __bf16 xp[KC_H * 32], hp[KC_H * 32];
    int tid = threadIdx.x, nt = blockDim.x;
    int lane = tid & 31, wave = tid >> 5, nwaves = nt >> 5;

    for (int i = tid; i < HDIM; i += nt) h[i] = 0.0f;
    __syncthreads();

    for (int t = 0; t < SLEN; ++t) {
        int tok = tokens[t];
        for (int i = tid; i < HDIM; i += nt) x[i] = embed[(long long)tok * HDIM + i];
        __syncthreads();
        build_xpack(x, xp, KC_H, tid, nt);
        build_xpack(h, hp, KC_H, tid, nt);
        __syncthreads();
        wave_gemv<4>(Wxp, xp, KC_H, NT_3H, 3 * HDIM, b3, gx, wave, nwaves, lane);
        wave_gemv<4>(Whp, hp, KC_H, NT_3H, 3 * HDIM, nullptr, gh, wave, nwaves, lane);
        __syncthreads();
        for (int i = tid; i < HDIM; i += nt) {
            float r = sigmoidf_(gx[i] + gh[i]);
            float z = sigmoidf_(gx[HDIM + i] + gh[HDIM + i]);
            float n = tanhf(gx[2 * HDIM + i] + r * gh[2 * HDIM + i]);
            float hn = (1.0f - z) * n + z * h[i];
            h[i] = hn;
            encOut[(long long)t * HDIM + i] = hn;
        }
        __syncthreads();
    }
}

// ---------------------------------------------------------------------------
// Decoder: one persistent workgroup, maxLen sequential steps with attention,
// output projection (bf16 Wo streamed from L2), softmax, and top-p sampling
// via threshold binary search + block scan (no global sort).
// ---------------------------------------------------------------------------
__global__ void __launch_bounds__(1024)
decoder_kernel(const float* __restrict__ embed,
               const __bf16* __restrict__ Wxp, const __bf16* __restrict__ Whp,
               const float* __restrict__ b3,
               const __bf16* __restrict__ Wcp, const __bf16* __restrict__ Wop,
               const float* __restrict__ bo,
               const float* __restrict__ encOut, const int* __restrict__ maxLenP,
               float* __restrict__ e_ws, float* __restrict__ out) {
    __shared__ float x[HDIM], h[HDIM], gx[3 * HDIM], gh[3 * HDIM];
    __shared__ float ctx[HDIM], comb[HDIM], cat[2 * HDIM];
    __shared__ float attn[SLEN];
    __shared__ float red[1024];
    __shared__ __align__(32) __bf16 xp[KC_2H * 32], hp[KC_H * 32];
    __shared__ int stok;

    int tid = threadIdx.x, nt = blockDim.x;
    int lane = tid & 31, wave = tid >> 5, nwaves = nt >> 5;
    int maxLen = maxLenP[0];
    if (maxLen < 1) maxLen = 1;
    if (maxLen > 4096) maxLen = 4096;

    for (int i = tid; i < HDIM; i += nt) h[i] = encOut[(long long)(SLEN - 1) * HDIM + i];
    if (tid == 0) stok = SOS_TOKEN;
    __syncthreads();

    const int CH = (VOCAB + 1023) / 1024; // per-thread chunk for scans

    for (int step = 0; step < maxLen; ++step) {
        // ---- GRU cell ----
        int tok = stok;
        for (int i = tid; i < HDIM; i += nt) x[i] = embed[(long long)tok * HDIM + i];
        __syncthreads();
        build_xpack(x, xp, KC_H, tid, nt);
        build_xpack(h, hp, KC_H, tid, nt);
        __syncthreads();
        wave_gemv<4>(Wxp, xp, KC_H, NT_3H, 3 * HDIM, b3, gx, wave, nwaves, lane);
        wave_gemv<4>(Whp, hp, KC_H, NT_3H, 3 * HDIM, nullptr, gh, wave, nwaves, lane);
        __syncthreads();
        for (int i = tid; i < HDIM; i += nt) {
            float r = sigmoidf_(gx[i] + gh[i]);
            float z = sigmoidf_(gx[HDIM + i] + gh[HDIM + i]);
            float n = tanhf(gx[2 * HDIM + i] + r * gh[2 * HDIM + i]);
            h[i] = (1.0f - z) * n + z * h[i];
        }
        __syncthreads();

        // ---- attention over encoder states ----
        if (tid < SLEN) {
            float d = 0.0f;
            const float* er = encOut + (long long)tid * HDIM;
            for (int k = 0; k < HDIM; ++k) d += er[k] * h[k];
            attn[tid] = d;
        }
        __syncthreads();
        if (tid == 0) {
            float m = attn[0];
            for (int s2 = 1; s2 < SLEN; ++s2) m = fmaxf(m, attn[s2]);
            float ss = 0.0f;
            for (int s2 = 0; s2 < SLEN; ++s2) { attn[s2] = __expf(attn[s2] - m); ss += attn[s2]; }
            float inv = 1.0f / ss;
            for (int s2 = 0; s2 < SLEN; ++s2) attn[s2] *= inv;
        }
        __syncthreads();
        for (int i = tid; i < HDIM; i += nt) {
            float c = 0.0f;
            for (int s2 = 0; s2 < SLEN; ++s2) c += attn[s2] * encOut[(long long)s2 * HDIM + i];
            ctx[i] = c;
            cat[i] = h[i];
            cat[HDIM + i] = c;
        }
        __syncthreads();

        // ---- comb = tanh(cat @ Wc) ----
        build_xpack(cat, xp, KC_2H, tid, nt);
        __syncthreads();
        wave_gemv<4>(Wcp, xp, KC_2H, NT_H, HDIM, nullptr, gx, wave, nwaves, lane);
        __syncthreads();
        for (int i = tid; i < HDIM; i += nt) comb[i] = tanhf(gx[i]);
        __syncthreads();

        // ---- logits = comb @ Wo + bo (the big L2-resident bf16 stream) ----
        build_xpack(comb, xp, KC_H, tid, nt);
        __syncthreads();
        wave_gemv<4>(Wop, xp, KC_H, NT_WO, VOCAB, bo, e_ws, wave, nwaves, lane);
        __syncthreads();

        // ---- softmax over V: keep unnormalized e_i = exp(l - max), e_max == 1 ----
        float lm = -3.0e38f;
        for (int i = tid; i < VOCAB; i += nt) lm = fmaxf(lm, e_ws[i]);
        float M = block_max(lm, red, tid, nt);
        float ls = 0.0f;
        for (int i = tid; i < VOCAB; i += nt) {
            float e = __expf(e_ws[i] - M);
            e_ws[i] = e;
            ls += e;
        }
        float Ssum = block_sum(ls, red, tid, nt);

        // ---- top-p threshold via binary search on tau in [0, 1] ----
        float target_keep = TOPP * Ssum;
        float lo = 0.0f, hi = 1.0f;
        for (int it = 0; it < 25; ++it) {
            float mid = 0.5f * (lo + hi);
            float pm = 0.0f;
            for (int i = tid; i < VOCAB; i += nt) {
                float e = e_ws[i];
                pm += (e >= mid) ? e : 0.0f;
            }
            float mass = block_sum(pm, red, tid, nt);
            if (mass >= target_keep) lo = mid; else hi = mid;
        }
        float tau = lo;
        float pm = 0.0f;
        for (int i = tid; i < VOCAB; i += nt) {
            float e = e_ws[i];
            pm += (e >= tau) ? e : 0.0f;
        }
        float keepMass = block_sum(pm, red, tid, nt);

        // ---- sample within the nucleus: block scan + crossing search ----
        float u = rand01(0x2A2A2A2Aull * 131ull + (unsigned long long)step);
        float target = fminf(u * keepMass, keepMass * 0.9999994f);

        int base = tid * CH;
        float local = 0.0f;
        for (int i = base; i < base + CH && i < VOCAB; ++i) {
            float e = e_ws[i];
            local += (e >= tau) ? e : 0.0f;
        }
        red[tid] = local;
        __syncthreads();
        for (int off = 1; off < nt; off <<= 1) {   // Hillis-Steele inclusive scan
            float v = (tid >= off) ? red[tid - off] : 0.0f;
            __syncthreads();
            red[tid] += v;
            __syncthreads();
        }
        float incl = red[tid];
        float excl = incl - local;
        if (tid == 0) stok = 0;
        __syncthreads();
        if (local > 0.0f && target >= excl && target < incl) {
            float cum = excl;
            for (int i = base; i < base + CH && i < VOCAB; ++i) {
                float e = e_ws[i];
                if (e >= tau) {
                    cum += e;
                    if (cum > target) { stok = i; break; }
                }
            }
        }
        __syncthreads();

        if (tid == 0) {
            int nt_ = stok;
            out[step] = (float)nt_;
            out[maxLen + step] = e_ws[nt_] / Ssum;
        }
        __syncthreads();
    }
}

// ---------------------------------------------------------------------------
// Launch
// ---------------------------------------------------------------------------
extern "C" void kernel_launch(void* const* d_in, const int* in_sizes, int n_in,
                              void* d_out, int out_size, void* d_ws, size_t ws_size,
                              hipStream_t stream) {
    (void)in_sizes; (void)n_in; (void)out_size; (void)ws_size;

    const int*   inputSeq = (const int*)d_in[0];
    const int*   maxLenP  = (const int*)d_in[1];
    const float* embed    = (const float*)d_in[2];
    const float* enc_Wx   = (const float*)d_in[3];
    const float* enc_Wh   = (const float*)d_in[4];
    const float* enc_b    = (const float*)d_in[5];
    const float* dec_Wx   = (const float*)d_in[6];
    const float* dec_Wh   = (const float*)d_in[7];
    const float* dec_b    = (const float*)d_in[8];
    const float* Wc       = (const float*)d_in[9];
    const float* Wo       = (const float*)d_in[10];
    const float* bo       = (const float*)d_in[11];
    float* out = (float*)d_out;

    char* base = (char*)d_ws;
    size_t off = 0;
    auto take = [&](size_t bytes) -> void* {
        void* p = base + off;
        off += bytes;
        off = (off + 255) & ~(size_t)255;
        return p;
    };

    const size_t sz3H = (size_t)NT_3H * KC_H * 512;   // packed elems for 1024x3072
    const size_t szWc = (size_t)NT_H * KC_2H * 512;   // packed elems for 2048x1024
    const size_t szWo = (size_t)NT_WO * KC_H * 512;   // packed elems for 1024x50272

    __bf16* encWxp = (__bf16*)take(sz3H * 2);
    __bf16* encWhp = (__bf16*)take(sz3H * 2);
    __bf16* decWxp = (__bf16*)take(sz3H * 2);
    __bf16* decWhp = (__bf16*)take(sz3H * 2);
    __bf16* Wcp    = (__bf16*)take(szWc * 2);
    __bf16* Wop    = (__bf16*)take(szWo * 2);
    float*  encOut = (float*)take((size_t)SLEN * HDIM * 4);
    float*  e_ws   = (float*)take((size_t)VPAD * 4);

    auto launch_pack = [&](const float* src, __bf16* dst, int Ktot, int Ntot, int ntiles) {
        int Kc = (Ktot + 31) >> 5;
        long long total = (long long)ntiles * Kc * 512;
        int blocks = (int)((total + 255) / 256);
        pack_bf16_kernel<<<blocks, 256, 0, stream>>>(src, dst, Ktot, Ntot, ntiles);
    };

    launch_pack(enc_Wx, encWxp, HDIM, 3 * HDIM, NT_3H);
    launch_pack(enc_Wh, encWhp, HDIM, 3 * HDIM, NT_3H);
    launch_pack(dec_Wx, decWxp, HDIM, 3 * HDIM, NT_3H);
    launch_pack(dec_Wh, decWhp, HDIM, 3 * HDIM, NT_3H);
    launch_pack(Wc, Wcp, 2 * HDIM, HDIM, NT_H);
    launch_pack(Wo, Wop, HDIM, VOCAB, NT_WO);

    encoder_kernel<<<1, 1024, 0, stream>>>(inputSeq, embed, encWxp, encWhp, enc_b, encOut);
    decoder_kernel<<<1, 1024, 0, stream>>>(embed, decWxp, decWhp, dec_b, Wcp, Wop, bo,
                                           encOut, maxLenP, e_ws, out);
}
